// ResidualBlock_8014408974759
// MI455X (gfx1250) — compile-verified, matrix operand order checked by reading, not run on hardware
//
#include <hip/hip_runtime.h>
#include <hip/hip_bf16.h>
#include <math.h>

// ---------- problem constants (match reference) ----------
static constexpr int BATCH = 2, SEQ = 4096, DIN = 1024, DM = 2048, KC = 4;
static constexpr int MROWS = BATCH * SEQ;         // 8192
static constexpr float EPS = 1e-5f;

// ---------- CDNA5 WMMA vector types ----------
typedef __attribute__((ext_vector_type(16))) __bf16 v16bf;
typedef __attribute__((ext_vector_type(8)))  __bf16 v8bf;
typedef __attribute__((ext_vector_type(8)))  float  v8f;
typedef int b128i __attribute__((vector_size(16)));   // 128-bit payload type for async LDS copies

#define GLB_AS __attribute__((address_space(1)))
#define LDS_AS __attribute__((address_space(3)))

#if __has_builtin(__builtin_amdgcn_global_load_async_to_lds_b128)
#define HAVE_ASYNC_LDS 1
#else
#define HAVE_ASYNC_LDS 0
#endif

__device__ __forceinline__ float silu_f(float x) { return x / (1.0f + __expf(-x)); }

// 16-byte global -> LDS copy; async (no VGPR round trip) when available.
__device__ __forceinline__ void cp_b128(void* lds, const void* g)
{
#if HAVE_ASYNC_LDS
    __builtin_amdgcn_global_load_async_to_lds_b128(
        (GLB_AS b128i*)g, (LDS_AS b128i*)lds, 0, 0);
#else
    *(v8bf*)lds = *(const v8bf*)g;
#endif
}

__device__ __forceinline__ void wait_async_all()
{
#if HAVE_ASYNC_LDS
#if __has_builtin(__builtin_amdgcn_s_wait_asynccnt)
    __builtin_amdgcn_s_wait_asynccnt(0);
#else
    asm volatile("s_wait_asynccnt 0x0" ::: "memory");
#endif
#endif
}

// =====================================================================
// 0) convert+transpose fp32 weight [Kd][N] -> bf16 [N][Kd]
// =====================================================================
__global__ __launch_bounds__(256) void cvt_transpose_kernel(
    const float* __restrict__ w, __bf16* __restrict__ wt, int Kd, int N)
{
    int idx = blockIdx.x * 256 + threadIdx.x;       // over N*Kd
    int n = idx / Kd;
    int k = idx - n * Kd;
    wt[idx] = (__bf16)w[(size_t)k * N + n];
}

// =====================================================================
// 1) RMSNorm over DIN, output bf16 activations
// =====================================================================
__global__ __launch_bounds__(256) void rmsnorm_kernel(
    const float* __restrict__ x, const float* __restrict__ w,
    __bf16* __restrict__ h)
{
    const int row = blockIdx.x;
    const float* xr = x + (size_t)row * DIN;
    float vals[DIN / 256];
    float ss = 0.f;
#pragma unroll
    for (int i = 0; i < DIN / 256; ++i) {
        float v = xr[threadIdx.x + i * 256];
        vals[i] = v;
        ss += v * v;
    }
#pragma unroll
    for (int off = 16; off > 0; off >>= 1) ss += __shfl_xor(ss, off, 32);
    __shared__ float red[8];
    int wid = threadIdx.x >> 5, lane = threadIdx.x & 31;
    if (lane == 0) red[wid] = ss;
    __syncthreads();
    float tot = 0.f;
#pragma unroll
    for (int i = 0; i < 8; ++i) tot += red[i];
    const float scale = rsqrtf(tot * (1.0f / DIN) + EPS);
    __bf16* hr = h + (size_t)row * DIN;
#pragma unroll
    for (int i = 0; i < DIN / 256; ++i) {
        int d = threadIdx.x + i * 256;
        hr[d] = (__bf16)(vals[i] * scale * w[d]);
    }
}

// =====================================================================
// helper: assemble a 16x32 bf16 WMMA fragment from an LDS tile row
// (lanes 0-15: K 0..7 & 16..23 ; lanes 16-31: K 8..15 & 24..31)
// =====================================================================
__device__ __forceinline__ v16bf load_frag(const __bf16* rowp, int hi)
{
    union { v16bf v; v8bf h[2]; } u;
    u.h[0] = *(const v8bf*)(rowp + hi * 8);
    u.h[1] = *(const v8bf*)(rowp + 16 + hi * 8);
    return u.v;
}

// =====================================================================
// 2) Dual GEMM: xi = H x Win ; sres = silu(H x Wres)   (M=8192,K=1024,N=2048)
//    Double-buffered LDS, async global->LDS staging, BK=32.
//    Block tile 128x64, 8 waves in 4x2 grid, 32x32 wave tiles.
// =====================================================================
static constexpr int D_BM = 128, D_BN = 64, D_BK = 32, D_LDT = 40; // 80B rows

__device__ __forceinline__ void stage_dual(
    __bf16 (*As)[D_LDT], __bf16 (*Bs1)[D_LDT], __bf16 (*Bs2)[D_LDT],
    const __bf16* A, const __bf16* B1t, const __bf16* B2t,
    int rowBase, int colBase, int k0, int tid)
{
#pragma unroll
    for (int i = 0; i < 2; ++i) {
        int c = tid + i * 256;                    // 512 A chunks
        int r = c >> 2, cc = (c & 3) * 8;
        cp_b128(&As[r][cc], &A[(size_t)(rowBase + r) * DIN + k0 + cc]);
    }
    int r = tid >> 2, cc = (tid & 3) * 8;         // 256 chunks each B tile
    cp_b128(&Bs1[r][cc], &B1t[(size_t)(colBase + r) * DIN + k0 + cc]);
    cp_b128(&Bs2[r][cc], &B2t[(size_t)(colBase + r) * DIN + k0 + cc]);
}

__global__ __launch_bounds__(256) void gemm_dual_kernel(
    const __bf16* __restrict__ A,
    const __bf16* __restrict__ B1t,
    const __bf16* __restrict__ B2t,
    float*  __restrict__ xi,
    __bf16* __restrict__ sres)
{
    __shared__ __align__(16) __bf16 As [2][D_BM][D_LDT];
    __shared__ __align__(16) __bf16 Bs1[2][D_BN][D_LDT];
    __shared__ __align__(16) __bf16 Bs2[2][D_BN][D_LDT];

    const int tid  = threadIdx.x;
    const int wave = tid >> 5, lane = tid & 31;
    const int wm = wave >> 1, wn = wave & 1;        // 4 x 2 wave grid
    const int lr = lane & 15, hi = lane >> 4;
    const int rowBase = blockIdx.x * D_BM;
    const int colBase = blockIdx.y * D_BN;

    v8f c1[2][2], c2[2][2];
#pragma unroll
    for (int t = 0; t < 2; ++t)
#pragma unroll
        for (int u = 0; u < 2; ++u) { c1[t][u] = (v8f)0.0f; c2[t][u] = (v8f)0.0f; }

    constexpr int KT = DIN / D_BK;
    stage_dual(As[0], Bs1[0], Bs2[0], A, B1t, B2t, rowBase, colBase, 0, tid);

    for (int kt = 0; kt < KT; ++kt) {
        wait_async_all();            // own async copies into buf(kt) done
        __syncthreads();             // all waves' copies visible; prior reads drained
        if (kt + 1 < KT)
            stage_dual(As[(kt + 1) & 1], Bs1[(kt + 1) & 1], Bs2[(kt + 1) & 1],
                       A, B1t, B2t, rowBase, colBase, (kt + 1) * D_BK, tid);
        const int b = kt & 1;

        v16bf a[2], b1[2], b2[2];
#pragma unroll
        for (int t = 0; t < 2; ++t)
            a[t] = load_frag(&As[b][wm * 32 + t * 16 + lr][0], hi);
#pragma unroll
        for (int u = 0; u < 2; ++u) {
            b1[u] = load_frag(&Bs1[b][wn * 32 + u * 16 + lr][0], hi);
            b2[u] = load_frag(&Bs2[b][wn * 32 + u * 16 + lr][0], hi);
        }
#pragma unroll
        for (int t = 0; t < 2; ++t)
#pragma unroll
            for (int u = 0; u < 2; ++u) {
                c1[t][u] = __builtin_amdgcn_wmma_f32_16x16x32_bf16(
                    false, a[t], false, b1[u], (short)0, c1[t][u], false, false);
                c2[t][u] = __builtin_amdgcn_wmma_f32_16x16x32_bf16(
                    false, a[t], false, b2[u], (short)0, c2[t][u], false, false);
            }
    }

    // ---- epilogue: xi fp32 ; sres = bf16(silu(res)) ----
#pragma unroll
    for (int t = 0; t < 2; ++t)
#pragma unroll
        for (int u = 0; u < 2; ++u)
#pragma unroll
            for (int r = 0; r < 8; ++r) {
                int m = rowBase + wm * 32 + t * 16 + hi * 8 + r;
                int n = colBase + wn * 32 + u * 16 + lr;
                size_t o = (size_t)m * DM + n;
                xi[o]   = c1[t][u][r];
                sres[o] = (__bf16)silu_f(c2[t][u][r]);
            }
}

// =====================================================================
// 3) causal depthwise conv (K=4) + bias + SiLU, gated by silu(res)
//    Each thread: one channel x 8-long sequence strip (11 loads / 8 stores).
// =====================================================================
__global__ __launch_bounds__(256) void conv_gate_kernel(
    const float*  __restrict__ xi,
    const __bf16* __restrict__ sres,
    const float*  __restrict__ cw,    // [DM][KC]
    const float*  __restrict__ cb,    // [DM]
    __bf16* __restrict__ y)
{
    size_t idx  = (size_t)blockIdx.x * 256 + threadIdx.x; // over (MROWS/8)*DM
    int    m    = (int)(idx % DM);
    size_t strip = idx / DM;
    size_t row0 = strip * 8;                 // SEQ % 8 == 0: strip stays in one batch
    int    l0   = (int)(row0 % SEQ);

    const float w0 = cw[m * KC + 0], w1 = cw[m * KC + 1];
    const float w2 = cw[m * KC + 2], w3 = cw[m * KC + 3];
    const float bias = cb[m];

    float xv[8 + KC - 1];
#pragma unroll
    for (int j = 0; j < 8 + KC - 1; ++j) {
        int dl = l0 - (KC - 1) + j;
        xv[j] = (dl >= 0) ? xi[(row0 + (size_t)j - (size_t)(KC - 1)) * DM + m] : 0.0f;
    }
#pragma unroll
    for (int i = 0; i < 8; ++i) {
        float acc = bias + xv[i] * w0 + xv[i + 1] * w1 + xv[i + 2] * w2 + xv[i + 3] * w3;
        size_t o = (row0 + i) * DM + m;
        y[o] = (__bf16)(silu_f(acc) * (float)sres[o]);
    }
}

// =====================================================================
// 4) Output GEMM + residual: out = Y x Wout + x  (M=8192,K=2048,N=1024)
//    Double-buffered LDS, async staging. Block 128x128, 2x4 waves, 64x32 tiles.
// =====================================================================
static constexpr int O_BM = 128, O_BN = 128, O_BK = 32, O_LDT = 40;

__device__ __forceinline__ void stage_out(
    __bf16 (*As)[O_LDT], __bf16 (*Bs)[O_LDT],
    const __bf16* Y, const __bf16* Wt,
    int rowBase, int colBase, int k0, int tid)
{
#pragma unroll
    for (int i = 0; i < 2; ++i) {
        int c = tid + i * 256;
        int r = c >> 2, cc = (c & 3) * 8;
        cp_b128(&As[r][cc], &Y [(size_t)(rowBase + r) * DM + k0 + cc]);
        cp_b128(&Bs[r][cc], &Wt[(size_t)(colBase + r) * DM + k0 + cc]);
    }
}

__global__ __launch_bounds__(256) void gemm_out_kernel(
    const __bf16* __restrict__ Y,    // [M][DM]
    const __bf16* __restrict__ Wt,   // [DIN][DM]  (pre-transposed)
    const float*  __restrict__ x,    // [M][DIN]
    float* __restrict__ out)
{
    __shared__ __align__(16) __bf16 As[2][O_BM][O_LDT];
    __shared__ __align__(16) __bf16 Bs[2][O_BN][O_LDT];

    const int tid  = threadIdx.x;
    const int wave = tid >> 5, lane = tid & 31;
    const int wm = wave >> 2, wn = wave & 3;       // 2 x 4 wave grid
    const int lr = lane & 15, hi = lane >> 4;
    const int rowBase = blockIdx.x * O_BM;
    const int colBase = blockIdx.y * O_BN;

    v8f c[4][2];
#pragma unroll
    for (int t = 0; t < 4; ++t)
#pragma unroll
        for (int u = 0; u < 2; ++u) c[t][u] = (v8f)0.0f;

    constexpr int KT = DM / O_BK;
    stage_out(As[0], Bs[0], Y, Wt, rowBase, colBase, 0, tid);

    for (int kt = 0; kt < KT; ++kt) {
        wait_async_all();
        __syncthreads();
        if (kt + 1 < KT)
            stage_out(As[(kt + 1) & 1], Bs[(kt + 1) & 1],
                      Y, Wt, rowBase, colBase, (kt + 1) * O_BK, tid);
        const int b = kt & 1;

        v16bf a[4], bb[2];
#pragma unroll
        for (int t = 0; t < 4; ++t)
            a[t] = load_frag(&As[b][wm * 64 + t * 16 + lr][0], hi);
#pragma unroll
        for (int u = 0; u < 2; ++u)
            bb[u] = load_frag(&Bs[b][wn * 32 + u * 16 + lr][0], hi);

#pragma unroll
        for (int t = 0; t < 4; ++t)
#pragma unroll
            for (int u = 0; u < 2; ++u)
                c[t][u] = __builtin_amdgcn_wmma_f32_16x16x32_bf16(
                    false, a[t], false, bb[u], (short)0, c[t][u], false, false);
    }

#pragma unroll
    for (int t = 0; t < 4; ++t)
#pragma unroll
        for (int u = 0; u < 2; ++u)
#pragma unroll
            for (int r = 0; r < 8; ++r) {
                int m = rowBase + wm * 64 + t * 16 + hi * 8 + r;
                int n = colBase + wn * 32 + u * 16 + lr;
                size_t o = (size_t)m * DIN + n;
                out[o] = c[t][u][r] + x[o];
            }
}

// =====================================================================
// launcher
// =====================================================================
extern "C" void kernel_launch(void* const* d_in, const int* in_sizes, int n_in,
                              void* d_out, int out_size, void* d_ws, size_t ws_size,
                              hipStream_t stream)
{
    (void)in_sizes; (void)n_in; (void)out_size; (void)ws_size;
    const float* x      = (const float*)d_in[0];
    const float* norm_w = (const float*)d_in[1];
    const float* w_in   = (const float*)d_in[2];
    const float* w_res  = (const float*)d_in[3];
    const float* conv_w = (const float*)d_in[4];
    const float* conv_b = (const float*)d_in[5];
    const float* w_out  = (const float*)d_in[6];
    float* out = (float*)d_out;

    // workspace layout
    char* ws = (char*)d_ws;
    size_t off = 0;
    __bf16* h     = (__bf16*)(ws + off); off += (size_t)MROWS * DIN * 2;  // 16.8 MB
    __bf16* winT  = (__bf16*)(ws + off); off += (size_t)DM * DIN * 2;     //  4.2 MB
    __bf16* wresT = (__bf16*)(ws + off); off += (size_t)DM * DIN * 2;     //  4.2 MB
    __bf16* woutT = (__bf16*)(ws + off); off += (size_t)DIN * DM * 2;     //  4.2 MB
    float*  xi    = (float*) (ws + off); off += (size_t)MROWS * DM * 4;   // 67.1 MB
    __bf16* sres  = (__bf16*)(ws + off); off += (size_t)MROWS * DM * 2;   // 33.6 MB
    __bf16* y     = (__bf16*)(ws + off); off += (size_t)MROWS * DM * 2;   // 33.6 MB

    // 0) weight convert + transpose to N-major bf16
    cvt_transpose_kernel<<<(DM * DIN) / 256, 256, 0, stream>>>(w_in,  winT,  DIN, DM);
    cvt_transpose_kernel<<<(DM * DIN) / 256, 256, 0, stream>>>(w_res, wresT, DIN, DM);
    cvt_transpose_kernel<<<(DIN * DM) / 256, 256, 0, stream>>>(w_out, woutT, DM, DIN);

    // 1) RMSNorm -> bf16
    rmsnorm_kernel<<<MROWS, 256, 0, stream>>>(x, norm_w, h);

    // 2) dual projection GEMM (WMMA bf16, async double-buffered)
    gemm_dual_kernel<<<dim3(MROWS / D_BM, DM / D_BN), 256, 0, stream>>>(h, winT, wresT, xi, sres);

    // 3) causal depthwise conv + SiLU + gate
    conv_gate_kernel<<<(int)(((size_t)(MROWS / 8) * DM) / 256), 256, 0, stream>>>(
        xi, sres, conv_w, conv_b, y);

    // 4) output GEMM + residual (WMMA bf16, async double-buffered)
    gemm_out_kernel<<<dim3(MROWS / O_BM, DIN / O_BN), 256, 0, stream>>>(y, woutT, x, out);
}
